// SelfEdgeUpdateBlock_7739531067652
// MI455X (gfx1250) — matrix operand order repaired and checked by reading, not compiled, not run on hardware
//
#include <hip/hip_runtime.h>
#include <stdint.h>

typedef __attribute__((ext_vector_type(16))) _Float16 v16h;
typedef __attribute__((ext_vector_type(8)))  float    v8f;

#define MUL0 32
#define MUL1 16
#define DIM  80
#define NGRAPH 512
#define EPSN 1e-5f
#define EPB  128   // edges per block (8 waves x 16-edge M-tiles)

// path normalizations, precomputed (f64 literals -> f32)
#define A_SS   0.022097086912079608f   // 1/sqrt(2*32*32)
#define A_VV   0.025515518153991442f   // 1/(sqrt(2*16*16)*sqrt(3))
#define A_SV   0.03125f                // 1/sqrt(2*32*16)
#define INV0   0.17677669529663687f    // 1/sqrt(32)
#define INV1   0.25f                   // 1/sqrt(16)
#define INVSC0 0.04419417382415922f    // 1/sqrt(32*16)
#define INVSC1 0.0625f                 // 1/sqrt(16*16)

// LDS row pitches (halfs): pitch/2 words % 64 == 4 -> B-fragment lane stride = 4 banks
#define P520 520
#define P264 264

// workspace byte offsets (all 16B-aligned)
#define OFF_WT1   0u         // 2 K-halves x 48 rows x P520 halfs  (99840 B)
#define OFF_WT2   99840u     // 48 x P264 halfs                     (25344 B)
#define OFF_WTSV  125184u    // 16 x P520 halfs                     (16640 B)
#define OFF_WTVS  141824u    // 16 x P520 halfs                     (16640 B)
#define OFF_M0    158464u    // 16 x 32 x 32 f32                    (65536 B)
#define OFF_B0    224000u    // 16 x 32 f32                         ( 2048 B)
#define OFF_M1    226048u    // 16 x 16 x 16 f32                    (16384 B)
#define OFF_STATS 242432u    // NGRAPH x 34 f32                     (69632 B)
#define OFF_S4V4  312064u    // E x 80 f32

__device__ __forceinline__ v8f wmma16(v16h a, v16h b, v8f c) {
  return __builtin_amdgcn_wmma_f32_16x16x32_f16(false, a, false, b, (short)0, c, false, false);
}

// ---------------------------------------------------------------------------
// prep: scale+transpose TP weights to f16 tables; fuse cfconv/lin_post/species
// ---------------------------------------------------------------------------
__global__ __launch_bounds__(256) void prep_kernel(
    const float* __restrict__ Wss_s, const float* __restrict__ Wvv_s,
    const float* __restrict__ Wss_g, const float* __restrict__ Wvv_g,
    const float* __restrict__ Wsv,   const float* __restrict__ Wvs,
    const float* __restrict__ Wc0,   const float* __restrict__ Wc1,
    const float* __restrict__ Wp0,   const float* __restrict__ bp0,
    const float* __restrict__ Wp1,   const float* __restrict__ Wsc0,
    const float* __restrict__ Wsc1,  char* __restrict__ ws) {
  const int t = threadIdx.x;
  _Float16* wt1  = (_Float16*)(ws + OFF_WT1);
  _Float16* wt2  = (_Float16*)(ws + OFF_WT2);
  _Float16* wtsv = (_Float16*)(ws + OFF_WTSV);
  _Float16* wtvs = (_Float16*)(ws + OFF_WTVS);
  float* M0 = (float*)(ws + OFF_M0);
  float* B0 = (float*)(ws + OFF_B0);
  float* M1 = (float*)(ws + OFF_M1);

  // WT1[n][k] = concat(W_ss_s, W_ss_g)[u,v,n] * A_SS, k=u*32+v, split into two K-halves
  for (int i = t; i < 2 * 48 * P520; i += 256) {
    int half = i / (48 * P520), rem = i % (48 * P520);
    int n = rem / P520, k = rem % P520;
    float val = 0.f;
    if (k < 512) {
      int kg = half * 512 + k, u = kg >> 5, v = kg & 31;
      val = (n < 32) ? Wss_s[(u * 32 + v) * 32 + n] * A_SS
                     : Wss_g[(u * 32 + v) * 16 + (n - 32)] * A_SS;
    }
    wt1[i] = (_Float16)val;
  }
  // WT2[n][k] = concat(W_vv_s, W_vv_g)[u,v,n] * A_VV, k=u*16+v
  for (int i = t; i < 48 * P264; i += 256) {
    int n = i / P264, k = i % P264;
    float val = 0.f;
    if (k < 256) {
      int u = k >> 4, v = k & 15;
      val = (n < 32) ? Wvv_s[(u * 16 + v) * 32 + n] * A_VV
                     : Wvv_g[(u * 16 + v) * 16 + (n - 32)] * A_VV;
    }
    wt2[i] = (_Float16)val;
  }
  // WTSV[n][k] = W_sv[u,v,n]*A_SV, k=u*16+v (u scalar, v vector)
  for (int i = t; i < 16 * P520; i += 256) {
    int n = i / P520, k = i % P520;
    float val = 0.f;
    if (k < 512) { int u = k >> 4, v = k & 15; val = Wsv[(u * 16 + v) * 16 + n] * A_SV; }
    wtsv[i] = (_Float16)val;
  }
  // WTVS[n][k] = W_vs[u,v,n]*A_SV, k=u*32+v (u vector, v scalar)
  for (int i = t; i < 16 * P520; i += 256) {
    int n = i / P520, k = i % P520;
    float val = 0.f;
    if (k < 512) { int u = k >> 5, v = k & 31; val = Wvs[(u * 32 + v) * 16 + n] * A_SV; }
    wtvs[i] = (_Float16)val;
  }

  // fused linear maps: Wcp = Wc @ Wp * INV^2
  __shared__ float sWcp0[32 * 32];
  __shared__ float sWcp1[16 * 16];
  for (int i = t; i < 1024; i += 256) {
    int u = i >> 5, w = i & 31;
    float acc = 0.f;
    for (int k = 0; k < 32; ++k) acc += Wc0[u * 32 + k] * Wp0[k * 32 + w];
    sWcp0[i] = acc * INV0 * INV0;
  }
  for (int i = t; i < 256; i += 256) {
    int u = i >> 4, w = i & 15;
    float acc = 0.f;
    for (int k = 0; k < 16; ++k) acc += Wc1[u * 16 + k] * Wp1[k * 16 + w];
    sWcp1[i] = acc * INV1 * INV1;
  }
  __syncthreads();
  // M0[spec][u][w] = Wcp0[u][w] + INVSC0 * sum_t Wcp0[u][t]*Wsc0[t,spec,w]
  for (int i = t; i < 16 * 1024; i += 256) {
    int spec = i >> 10, r = i & 1023, u = r >> 5, w = r & 31;
    float acc = 0.f;
    for (int k = 0; k < 32; ++k) acc += sWcp0[u * 32 + k] * Wsc0[(k * 16 + spec) * 32 + w];
    M0[i] = sWcp0[u * 32 + w] + INVSC0 * acc;
  }
  // B0[spec][w] = bp0[w] + INVSC0 * sum_t bp0[t]*Wsc0[t,spec,w]
  for (int i = t; i < 16 * 32; i += 256) {
    int spec = i >> 5, w = i & 31;
    float acc = 0.f;
    for (int k = 0; k < 32; ++k) acc += bp0[k] * Wsc0[(k * 16 + spec) * 32 + w];
    B0[i] = bp0[w] + INVSC0 * acc;
  }
  // M1[spec][u][w] = Wcp1[u][w] + INVSC1 * sum_t Wcp1[u][t]*Wsc1[t,spec,w]
  for (int i = t; i < 16 * 256; i += 256) {
    int spec = i >> 8, r = i & 255, u = r >> 4, w = r & 15;
    float acc = 0.f;
    for (int k = 0; k < 16; ++k) acc += sWcp1[u * 16 + k] * Wsc1[(k * 16 + spec) * 16 + w];
    M1[i] = sWcp1[u * 16 + w] + INVSC1 * acc;
  }
}

// ---------------------------------------------------------------------------
// tp: WMMA tensor product + gate + fused (cfconv/lin_post/species) epilogue
// ---------------------------------------------------------------------------
__global__ __launch_bounds__(256) void tp_kernel(
    const float* __restrict__ node_fea, const float* __restrict__ one_hot,
    const int* __restrict__ edge_index, char* __restrict__ ws, int E) {
  __shared__ alignas(16) _Float16 sWT1[48 * P520];   // one K-half at a time
  __shared__ alignas(16) _Float16 sWT2[48 * P264];
  __shared__ alignas(16) _Float16 sWSV[16 * P520];
  __shared__ alignas(16) _Float16 sWVS[16 * P520];
  __shared__ float stage[EPB][97];       // in: s[0..31], v[32+u*3+m]; out: s0|gt|v0

  const int tid = threadIdx.x;
  const int e0 = blockIdx.x * EPB;
  const int* idx0 = edge_index;          // row 0 of [2,E]
  const uint4* gWT1 = (const uint4*)(ws + OFF_WT1);

  // cooperative LDS fills (128-bit copies)
  {
    uint4* d1 = (uint4*)sWT1;
    for (int i = tid; i < (48 * P520) / 8; i += 256) d1[i] = gWT1[i];
    uint4* d2 = (uint4*)sWT2;
    const uint4* g2 = (const uint4*)(ws + OFF_WT2);
    for (int i = tid; i < (48 * P264) / 8; i += 256) d2[i] = g2[i];
    uint4* d3 = (uint4*)sWSV;
    const uint4* g3 = (const uint4*)(ws + OFF_WTSV);
    for (int i = tid; i < (16 * P520) / 8; i += 256) d3[i] = g3[i];
    uint4* d4 = (uint4*)sWVS;
    const uint4* g4 = (const uint4*)(ws + OFF_WTVS);
    for (int i = tid; i < (16 * P520) / 8; i += 256) d4[i] = g4[i];
    // edge features: float4 global loads, b32 LDS stores (pitch-97 rows are odd on purpose)
    for (int i = tid; i < EPB * (DIM / 4); i += 256) {
      int e = i / (DIM / 4), q = i % (DIM / 4);
      const float4 val = ((const float4*)(node_fea + (size_t)idx0[e0 + e] * DIM))[q];
      float* dst = &stage[e][q * 4];
      dst[0] = val.x; dst[1] = val.y; dst[2] = val.z; dst[3] = val.w;
    }
  }
  __syncthreads();

  const int lane = tid & 31;
  const int wv   = tid >> 5;
  const int r0   = wv * 16;
  const bool hi  = lane >= 16;
  const int nc   = lane & 15;           // output column / A row
  const int kA   = hi ? 8 : 0;          // A-layout K sub-offset per half-wave
  const float* __restrict__ row = &stage[r0 + nc][0];

  auto loadB = [&](const _Float16* WT, int pitch, int n, int kb) -> v16h {
    const _Float16* p = WT + n * pitch + kb + (hi ? 8 : 0);
    v16h b;
#pragma unroll
    for (int t2 = 0; t2 < 16; ++t2) b[t2] = p[t2 < 8 ? t2 : t2 + 8];
    return b;
  };
  auto buildA1 = [&](int kb) -> v16h {   // ss: k=u*32+v, u uniform per fragment
    const float su = row[kb >> 5];
    v16h a;
#pragma unroll
    for (int j = 0; j < 16; ++j) {
      int v = kA + (j < 8 ? j : j + 8);
      a[j] = (_Float16)(su * row[v]);
    }
    return a;
  };
  auto buildA2 = [&](int kb) -> v16h {   // vv dot: k=u*16+v
    v16h a;
#pragma unroll
    for (int j = 0; j < 16; ++j) {
      int k = kb + kA + (j < 8 ? j : j + 8);
      int u = k >> 4, v = k & 15;
      float d = row[32 + u * 3] * row[32 + v * 3] + row[33 + u * 3] * row[33 + v * 3] +
                row[34 + u * 3] * row[34 + v * 3];
      a[j] = (_Float16)d;
    }
    return a;
  };
  auto buildA3 = [&](int kb, int m) -> v16h {  // s_u * v_{v,m}: k=u*16+v
    v16h a;
#pragma unroll
    for (int j = 0; j < 16; ++j) {
      int k = kb + kA + (j < 8 ? j : j + 8);
      a[j] = (_Float16)(row[k >> 4] * row[32 + (k & 15) * 3 + m]);
    }
    return a;
  };
  auto buildA4 = [&](int kb, int m) -> v16h {  // v_{u,m} * s_v: k=u*32+v
    const float vu = row[32 + (kb >> 5) * 3 + m];
    v16h a;
#pragma unroll
    for (int j = 0; j < 16; ++j) {
      int v = kA + (j < 8 ? j : j + 8);
      a[j] = (_Float16)(vu * row[v]);
    }
    return a;
  };

  v8f accS0 = {}, accS1 = {}, accG = {};
  // ss path, K half 0
  for (int kb = 0; kb < 512; kb += 32) {
    v16h a = buildA1(kb);
    accS0 = wmma16(a, loadB(sWT1, P520, nc, kb), accS0);
    accS1 = wmma16(a, loadB(sWT1, P520, nc + 16, kb), accS1);
    accG  = wmma16(a, loadB(sWT1, P520, nc + 32, kb), accG);
  }
  __syncthreads();
  {
    uint4* d1 = (uint4*)sWT1;
    const uint4* g1h = gWT1 + (48 * P520) / 8;
    for (int i = tid; i < (48 * P520) / 8; i += 256) d1[i] = g1h[i];
  }
  __syncthreads();
  // ss path, K half 1
  for (int kb = 512; kb < 1024; kb += 32) {
    v16h a = buildA1(kb);
    int kl = kb - 512;
    accS0 = wmma16(a, loadB(sWT1, P520, nc, kl), accS0);
    accS1 = wmma16(a, loadB(sWT1, P520, nc + 16, kl), accS1);
    accG  = wmma16(a, loadB(sWT1, P520, nc + 32, kl), accG);
  }
  // vv path (accumulates into same C: path scales pre-folded into weights)
  for (int kb = 0; kb < 256; kb += 32) {
    v16h a = buildA2(kb);
    accS0 = wmma16(a, loadB(sWT2, P264, nc, kb), accS0);
    accS1 = wmma16(a, loadB(sWT2, P264, nc + 16, kb), accS1);
    accG  = wmma16(a, loadB(sWT2, P264, nc + 32, kb), accG);
  }
  // vector outputs: sv + vs paths per Cartesian m
  v8f accV[3] = {};
#pragma unroll
  for (int m = 0; m < 3; ++m) {
    for (int kb = 0; kb < 512; kb += 32) {
      v16h a = buildA3(kb, m);
      accV[m] = wmma16(a, loadB(sWSV, P520, nc, kb), accV[m]);
    }
    for (int kb = 0; kb < 512; kb += 32) {
      v16h a = buildA4(kb, m);
      accV[m] = wmma16(a, loadB(sWVS, P520, nc, kb), accV[m]);
    }
  }

  // redistribute WMMA C-layout (lane=col, vgpr=row) into stage rows (own rows only)
  {
    const int mb = r0 + (hi ? 8 : 0);
#pragma unroll
    for (int j = 0; j < 8; ++j) {
      float* orow = &stage[mb + j][0];
      orow[nc]      = accS0[j];
      orow[16 + nc] = accS1[j];
      orow[32 + nc] = accG[j];
      orow[48 + nc * 3 + 0] = accV[0][j];
      orow[48 + nc * 3 + 1] = accV[1][j];
      orow[48 + nc * 3 + 2] = accV[2][j];
    }
  }
  __syncthreads();

  // epilogue: gate + fused species-dependent linear maps; 2 threads per edge
  {
    const float* M0 = (const float*)(ws + OFF_M0);
    const float* B0 = (const float*)(ws + OFF_B0);
    const float* M1 = (const float*)(ws + OFF_M1);
    float* s4v4 = (float*)(ws + OFF_S4V4);

    const int e = tid >> 1;
    const int halfo = tid & 1;
    const int eg = e0 + e;
    const float* srow = &stage[e][0];

    // species from one-hot (float4 reads)
    int spec = 0;
    {
      const float4* oh4 = (const float4*)(one_hot + (size_t)eg * 16);
#pragma unroll
      for (int q = 0; q < 4; ++q) {
        float4 o = oh4[q];
        spec += (o.x > 0.5f) ? (q * 4 + 0) : 0;
        spec += (o.y > 0.5f) ? (q * 4 + 1) : 0;
        spec += (o.z > 0.5f) ? (q * 4 + 2) : 0;
        spec += (o.w > 0.5f) ? (q * 4 + 3) : 0;
      }
    }

    float s1[32];
#pragma unroll
    for (int u = 0; u < 32; ++u) { float x = srow[u]; s1[u] = x / (1.f + __expf(-x)); }
    float v1[16][3];
#pragma unroll
    for (int u = 0; u < 16; ++u) {
      float gte = 1.f / (1.f + __expf(-srow[32 + u]));
      v1[u][0] = srow[48 + u * 3 + 0] * gte;
      v1[u][1] = srow[48 + u * 3 + 1] * gte;
      v1[u][2] = srow[48 + u * 3 + 2] * gte;
    }
    float* out = s4v4 + (size_t)eg * DIM;

    // scalars: 16 outputs per thread-half, float4 weight rows
    float accS[16];
#pragma unroll
    for (int w = 0; w < 16; ++w) accS[w] = B0[spec * 32 + halfo * 16 + w];
    const float* M0s = M0 + spec * 1024 + halfo * 16;
    for (int u = 0; u < 32; ++u) {
      float su = s1[u];
      const float4* mrow = (const float4*)(M0s + u * 32);
#pragma unroll
      for (int q = 0; q < 4; ++q) {
        float4 mv = mrow[q];
        accS[q * 4 + 0] += su * mv.x;
        accS[q * 4 + 1] += su * mv.y;
        accS[q * 4 + 2] += su * mv.z;
        accS[q * 4 + 3] += su * mv.w;
      }
    }
    {
      float4* o4 = (float4*)(out + halfo * 16);
#pragma unroll
      for (int q = 0; q < 4; ++q)
        o4[q] = make_float4(accS[q * 4 + 0], accS[q * 4 + 1], accS[q * 4 + 2], accS[q * 4 + 3]);
    }

    // vectors: 8 outputs x 3 components per thread-half
    float accVo[8][3] = {};
    const float* M1s = M1 + spec * 256 + halfo * 8;
    for (int u = 0; u < 16; ++u) {
      const float4* mrow = (const float4*)(M1s + u * 16);
#pragma unroll
      for (int q = 0; q < 2; ++q) {
        float4 mv = mrow[q];
        float mm[4] = {mv.x, mv.y, mv.z, mv.w};
#pragma unroll
        for (int j = 0; j < 4; ++j) {
          int w = q * 4 + j;
          accVo[w][0] += v1[u][0] * mm[j];
          accVo[w][1] += v1[u][1] * mm[j];
          accVo[w][2] += v1[u][2] * mm[j];
        }
      }
    }
    {
      float tmp[24];
#pragma unroll
      for (int w = 0; w < 8; ++w) {
        tmp[w * 3 + 0] = accVo[w][0];
        tmp[w * 3 + 1] = accVo[w][1];
        tmp[w * 3 + 2] = accVo[w][2];
      }
      float4* o4 = (float4*)(out + 32 + halfo * 24);
#pragma unroll
      for (int q = 0; q < 6; ++q)
        o4[q] = make_float4(tmp[q * 4 + 0], tmp[q * 4 + 1], tmp[q * 4 + 2], tmp[q * 4 + 3]);
    }
  }
}

// ---------------------------------------------------------------------------
// stats: per-graph channel means + scalar/vector RMS (one block per graph)
// ---------------------------------------------------------------------------
__global__ __launch_bounds__(256) void stats_kernel(
    const float* __restrict__ s4v4, const int* __restrict__ edge_index,
    const int* __restrict__ batch, float* __restrict__ stats, int E) {
  const int g = blockIdx.x;
  const int* idx0 = edge_index;
  // binary search [lo, hic) over sorted seg = batch[idx0[e]]
  int lo, hic;
  {
    int a = 0, b = E;
    while (a < b) { int m = (a + b) >> 1; if (batch[idx0[m]] < g) a = m + 1; else b = m; }
    lo = a;
    b = E;
    while (a < b) { int m = (a + b) >> 1; if (batch[idx0[m]] < g + 1) a = m + 1; else b = m; }
    hic = a;
  }
  float sumS[32];
#pragma unroll
  for (int c = 0; c < 32; ++c) sumS[c] = 0.f;
  float sumS2 = 0.f, sumV2 = 0.f;
  for (int e = lo + (int)threadIdx.x; e < hic; e += 256) {
    const float4* p4 = (const float4*)(s4v4 + (size_t)e * DIM);
#pragma unroll
    for (int q = 0; q < 8; ++q) {
      float4 x = p4[q];
      sumS[q * 4 + 0] += x.x; sumS[q * 4 + 1] += x.y;
      sumS[q * 4 + 2] += x.z; sumS[q * 4 + 3] += x.w;
      sumS2 += x.x * x.x + x.y * x.y + x.z * x.z + x.w * x.w;
    }
#pragma unroll
    for (int q = 8; q < 20; ++q) {
      float4 x = p4[q];
      sumV2 += x.x * x.x + x.y * x.y + x.z * x.z + x.w * x.w;
    }
  }
  __shared__ float red[34];
  if (threadIdx.x < 34) red[threadIdx.x] = 0.f;
  __syncthreads();
#pragma unroll
  for (int c = 0; c < 32; ++c) atomicAdd(&red[c], sumS[c]);
  atomicAdd(&red[32], sumS2);
  atomicAdd(&red[33], sumV2);
  __syncthreads();
  if (threadIdx.x == 0) {
    int n = hic - lo;
    float cnt = (float)(n > 0 ? n : 1);
    float inv = 1.f / cnt;
    float* st = stats + g * 34;
    float musq = 0.f;
    for (int c = 0; c < 32; ++c) { float mu = red[c] * inv; st[c] = mu; musq += mu * mu; }
    float varS = fmaxf((red[32] - cnt * musq) * (inv / 32.f), 0.f);
    st[32] = rsqrtf(varS + EPSN);
    float varV = red[33] * (inv / 48.f);
    st[33] = rsqrtf(varV + EPSN);
  }
}

// ---------------------------------------------------------------------------
// apply: layernorm (scalar mean-sub + per-block RMS) + affine + skip
// one float4 (4 channels) per thread -> pure b128 streaming
// ---------------------------------------------------------------------------
__global__ __launch_bounds__(256) void apply_kernel(
    const float* __restrict__ s4v4, const float* __restrict__ edge_fea,
    const int* __restrict__ edge_index, const int* __restrict__ batch,
    const float* __restrict__ stats, const float* __restrict__ g0,
    const float* __restrict__ b0, const float* __restrict__ g1,
    float* __restrict__ out, int E) {
  int i = blockIdx.x * 256 + threadIdx.x;          // index over float4s
  const int NQ = DIM / 4;                          // 20 float4s per edge
  if (i >= E * NQ) return;
  int e = i / NQ, q = i % NQ;
  int g = batch[edge_index[e]];
  const float* st = stats + g * 34;
  float4 x = ((const float4*)s4v4)[i];
  float4 f = ((const float4*)edge_fea)[i];
  float r[4] = {x.x, x.y, x.z, x.w};
  float ff[4] = {f.x, f.y, f.z, f.w};
  float o[4];
#pragma unroll
  for (int j = 0; j < 4; ++j) {
    int c = q * 4 + j;
    float val;
    if (c < 32) {
      val = (r[j] - st[c]) * st[32] * g0[c] + b0[c];
    } else {
      int u = (c - 32) / 3;
      val = r[j] * st[33] * g1[u];
    }
    o[j] = ff[j] + val;
  }
  ((float4*)out)[i] = make_float4(o[0], o[1], o[2], o[3]);
}

// ---------------------------------------------------------------------------
extern "C" void kernel_launch(void* const* d_in, const int* in_sizes, int n_in,
                              void* d_out, int out_size, void* d_ws, size_t ws_size,
                              hipStream_t stream) {
  (void)n_in; (void)out_size; (void)ws_size;
  const float* node_fea = (const float*)d_in[0];
  const float* one_hot  = (const float*)d_in[1];
  const float* edge_fea = (const float*)d_in[3];
  const int*   edge_index = (const int*)d_in[5];
  const int*   batch   = (const int*)d_in[6];
  const float* Wss_s = (const float*)d_in[7];
  const float* Wvv_s = (const float*)d_in[8];
  const float* Wss_g = (const float*)d_in[9];
  const float* Wvv_g = (const float*)d_in[10];
  const float* Wsv   = (const float*)d_in[11];
  const float* Wvs   = (const float*)d_in[12];
  const float* Wc0   = (const float*)d_in[13];
  const float* Wc1   = (const float*)d_in[14];
  const float* Wp0   = (const float*)d_in[15];
  const float* bp0   = (const float*)d_in[16];
  const float* Wp1   = (const float*)d_in[17];
  const float* Wsc0  = (const float*)d_in[18];
  const float* Wsc1  = (const float*)d_in[19];
  const float* g0    = (const float*)d_in[20];
  const float* b0    = (const float*)d_in[21];
  const float* g1    = (const float*)d_in[22];
  char* ws = (char*)d_ws;
  const int E = in_sizes[3] / DIM;

  prep_kernel<<<1, 256, 0, stream>>>(Wss_s, Wvv_s, Wss_g, Wvv_g, Wsv, Wvs, Wc0, Wc1,
                                     Wp0, bp0, Wp1, Wsc0, Wsc1, ws);
  tp_kernel<<<E / EPB, 256, 0, stream>>>(node_fea, one_hot, edge_index, ws, E);
  stats_kernel<<<NGRAPH, 256, 0, stream>>>((const float*)(ws + OFF_S4V4), edge_index, batch,
                                           (float*)(ws + OFF_STATS), E);
  apply_kernel<<<(E * (DIM / 4) + 255) / 256, 256, 0, stream>>>(
      (const float*)(ws + OFF_S4V4), edge_fea, edge_index, batch,
      (const float*)(ws + OFF_STATS), g0, b0, g1, (float*)d_out, E);
}
